// MambaBlock_69758858821940
// MI455X (gfx1250) — compile-verified, matrix operand order checked by reading, not run on hardware
//
#include <hip/hip_runtime.h>
#include <hip/hip_bf16.h>
#include <math.h>

typedef _Float16 f16;
typedef __attribute__((ext_vector_type(8)))  _Float16 v8h;
typedef __attribute__((ext_vector_type(16))) _Float16 v16h;
typedef __attribute__((ext_vector_type(8)))  float    v8f;

#define D_MODEL 1024
#define D_STATE 16
#define D_INNER 2048
#define BATCH   2
#define SEQ     1024
#define M_ROWS  (BATCH * SEQ)   // 2048 tokens
#define BTILE   (64 * 32)       // one staged B tile: 64 cols x 32 K halves

// ---------------------------------------------------------------- helpers
__device__ __forceinline__ float soft_clamp(float x, float lo, float hi, float sharp) {
    float c = 0.5f * (hi + lo);
    float r = 0.5f * (hi - lo);
    return tanhf((x - c) / (r + 1e-8f) * sharp) / sharp * r + c;
}
__device__ __forceinline__ float softplus_f(float x) {
    return (x > 20.f) ? x : log1pf(__expf(x));
}
__device__ __forceinline__ float sigmoid_f(float x) {
    return 1.f / (1.f + __expf(-x));
}

// ---------------------------------------------------------------- f32 -> f16
__global__ void cvt_f16_kernel(const float* __restrict__ src, f16* __restrict__ dst, int n) {
    int i = blockIdx.x * blockDim.x + threadIdx.x;
    int stride = gridDim.x * blockDim.x;
    for (; i < n; i += stride) dst[i] = (f16)src[i];
}

// ---------------------------------------------------------------- RMSNorm (one block per token row)
__global__ void rmsnorm_kernel(const float* __restrict__ x, const float* __restrict__ w,
                               f16* __restrict__ xn) {
    __shared__ float red[256];
    int m = blockIdx.x;
    const float4* xr = (const float4*)(x + (size_t)m * D_MODEL);
    float4 v = xr[threadIdx.x];
    red[threadIdx.x] = v.x * v.x + v.y * v.y + v.z * v.z + v.w * v.w;
    __syncthreads();
    for (int st = 128; st > 0; st >>= 1) {
        if (threadIdx.x < st) red[threadIdx.x] += red[threadIdx.x + st];
        __syncthreads();
    }
    float rstd = rsqrtf(red[0] * (1.f / (float)D_MODEL) + 1e-6f);
    int c = threadIdx.x * 4;
    f16* o = xn + (size_t)m * D_MODEL + c;
    o[0] = (f16)(v.x * rstd * w[c + 0]);
    o[1] = (f16)(v.y * rstd * w[c + 1]);
    o[2] = (f16)(v.z * rstd * w[c + 2]);
    o[3] = (f16)(v.w * rstd * w[c + 3]);
}

// ---------------------------------------------------------------- WMMA fragment load
// A (16x32 f16, MxK): lane<16 -> row r, K in {kb..kb+7, kb+16..kb+23};
//                     lane>=16 -> row r, K in {kb+8..kb+15, kb+24..kb+31}.
// B (32x16 KxN) fed from W stored (N,K) row-major => identical pattern with row = column n.
__device__ __forceinline__ v16h load_frag(const f16* p) {
    v8h lo = *(const v8h*)p;         // K offsets +0..7
    v8h hi = *(const v8h*)(p + 16);  // K offsets +16..23
    v16h f;
#pragma unroll
    for (int i = 0; i < 8; i++) { f[i] = lo[i]; f[i + 8] = hi[i]; }
    return f;
}

// ---------------------------------------------------------------- tiled WMMA GEMM (N % 64 == 0, K >= 64)
// Block = 8 waves -> 128(M) x 64(N) tile. B tile double-buffered through LDS; the global load
// for tile k+1 is issued before computing tile k so HBM/L2 latency hides behind 4 WMMAs.
// ep: 0 = store; 1 = dt epilogue (bias+softplus+soft_clamp); 2 = residual (+sigmoid scale)
__global__ void wmma_gemm_tiled(const f16* __restrict__ A, const f16* __restrict__ Bw,
                                float* __restrict__ C, int M, int N, int K,
                                int lda, int ldb, int ldc,
                                int ep, const float* __restrict__ ep_bias,
                                const float* __restrict__ ep_resid,
                                const float* __restrict__ ep_scale) {
    __shared__ f16 Bs[2 * BTILE];        // 8 KB double buffer
    int tid  = threadIdx.x;
    int lane = tid & 31;
    int wave = tid >> 5;
    int tileM = blockIdx.y * 128 + wave * 16;
    int tileN = blockIdx.x * 64;
    int hi = lane >> 4;
    int r  = lane & 15;

    const f16* arow = A + (size_t)(tileM + r) * lda + hi * 8;

    // cooperative B staging: 256 threads x 16B = 4 KB per K-step
    int bn = tid >> 2;                   // 0..63
    int bk = (tid & 3) * 8;              // 0,8,16,24
    const f16* bsrc = Bw + (size_t)(tileN + bn) * ldb + bk;
    int bofs = bn * 32 + bk;

    v8f acc[4] = {{}, {}, {}, {}};

    // prologue: stage tile 0, put tile 1 in flight
    v8h breg = *(const v8h*)bsrc;                 // global_load_b128
    *(v8h*)(Bs + bofs) = breg;                    // ds_store_b128
    breg = *(const v8h*)(bsrc + 32);              // tile 1 load issued early
    __builtin_prefetch(bsrc + 64, 0, 1);
    __syncthreads();                              // tile 0 visible

    int p = 0;
    for (int k = 0; k < K; k += 32) {
        v16h a = load_frag(arow + k);             // global_load_b128 x2
        __builtin_prefetch(arow + k + 64, 0, 1);
        const f16* buf = Bs + p * BTILE;
#pragma unroll
        for (int nn = 0; nn < 4; nn++) {
            v16h b = load_frag(buf + (nn * 16 + r) * 32 + hi * 8);  // ds_load_b128 x2
            acc[nn] = __builtin_amdgcn_wmma_f32_16x16x32_f16(
                          false, a, false, b, (short)0, acc[nn], false, false);
        }
        if (k + 32 < K) {
            __syncthreads();                      // all waves done reading buf[p^1]
            *(v8h*)(Bs + (p ^ 1) * BTILE + bofs) = breg;
            if (k + 64 < K) {
                breg = *(const v8h*)(bsrc + k + 64);   // next tile in flight
                __builtin_prefetch(bsrc + k + 96, 0, 1);
            }
            __syncthreads();                      // new buffer visible
            p ^= 1;
        }
    }

    // D layout: VGPR j -> M = tileM + j + 8*hi, N = subtile + r
    int rbase = tileM + hi * 8;
    float scale = (ep == 2) ? sigmoid_f(ep_scale[0]) * 0.5f : 0.f;
#pragma unroll
    for (int nn = 0; nn < 4; nn++) {
        int col = tileN + nn * 16 + r;
        float bias = (ep == 1) ? ep_bias[col] : 0.f;
#pragma unroll
        for (int j = 0; j < 8; j++) {
            int row = rbase + j;
            float v = acc[nn][j];
            if (ep == 1) {
                v = softplus_f(v + bias);
                v = soft_clamp(v, 0.001f, 0.1f, 0.5f);
            } else if (ep == 2) {
                v = ep_resid[(size_t)row * ldc + col] + scale * v;
            }
            C[(size_t)row * ldc + col] = v;
        }
    }
}

// ---------------------------------------------------------------- narrow WMMA GEMM (N == 16)
__global__ void wmma_gemm_narrow(const f16* __restrict__ A, const f16* __restrict__ Bw,
                                 float* __restrict__ C, int M, int K,
                                 int lda, int ldb, int ldc) {
    int lane = threadIdx.x & 31;
    int wave = threadIdx.x >> 5;
    int tileM = blockIdx.y * 128 + wave * 16;
    int hi = lane >> 4;
    int r  = lane & 15;
    const f16* arow = A  + (size_t)(tileM + r) * lda + hi * 8;
    const f16* brow = Bw + (size_t)r * ldb + hi * 8;

    v8f acc = {};
    for (int k = 0; k < K; k += 32) {
        v16h a = load_frag(arow + k);
        v16h b = load_frag(brow + k);
        __builtin_prefetch(arow + k + 64, 0, 1);
        __builtin_prefetch(brow + k + 64, 0, 1);
        acc = __builtin_amdgcn_wmma_f32_16x16x32_f16(
                  false, a, false, b, (short)0, acc, false, false);
    }
    int col   = r;
    int rbase = tileM + hi * 8;
#pragma unroll
    for (int j = 0; j < 8; j++)
        C[(size_t)(rbase + j) * ldc + col] = acc[j];
}

// ---------------------------------------------------------------- depthwise causal conv + SiLU
__global__ void conv_silu_kernel(const float* __restrict__ proj, const float* __restrict__ cw,
                                 const float* __restrict__ cb, float* __restrict__ xc_f,
                                 f16* __restrict__ xc_h) {
    int i = blockIdx.x * blockDim.x + threadIdx.x;
    if (i >= M_ROWS * D_INNER) return;
    int d = i % D_INNER;
    int l = (i / D_INNER) % SEQ;
    int b = i / (D_INNER * SEQ);
    float acc = cb[d];
#pragma unroll
    for (int j = 0; j < 4; j++) {
        int ls = l - 3 + j;
        if (ls >= 0)
            acc += proj[(size_t)(b * SEQ + ls) * (2 * D_INNER) + d] * cw[d * 4 + j];
    }
    float v = acc * sigmoid_f(acc);     // SiLU
    xc_f[i] = v;
    xc_h[i] = (f16)v;
}

// ---------------------------------------------------------------- selective-scan (one block per batch)
__global__ void scan_kernel(const float* __restrict__ dt, const float* __restrict__ xc,
                            const float* __restrict__ Bp, const float* __restrict__ Cp,
                            const float* __restrict__ A_log, const float* __restrict__ Dp,
                            float* __restrict__ xssm) {
    __shared__ float red[1024];
    int b   = blockIdx.x;
    int tid = threadIdx.x;
    int d0  = tid * 2;

    float A0[D_STATE], A1[D_STATE], h0[D_STATE], h1[D_STATE];
#pragma unroll
    for (int n = 0; n < D_STATE; n++) {
        A0[n] = -__expf(soft_clamp(A_log[(size_t)d0 * D_STATE + n], -5.f, 5.f, 0.5f));
        A1[n] = -__expf(soft_clamp(A_log[(size_t)(d0 + 1) * D_STATE + n], -5.f, 5.f, 0.5f));
        h0[n] = 0.f; h1[n] = 0.f;
    }
    float Dp0 = Dp[d0], Dp1 = Dp[d0 + 1];

    for (int t = 0; t < SEQ; t++) {
        size_t m = (size_t)b * SEQ + t;
        float dt0 = dt[m * D_INNER + d0], dt1 = dt[m * D_INNER + d0 + 1];
        float x0  = xc[m * D_INNER + d0], x1  = xc[m * D_INNER + d0 + 1];
        const float* Br = Bp + m * D_STATE;
        const float* Cr = Cp + m * D_STATE;

#pragma unroll
        for (int n = 0; n < D_STATE; n++) {
            float Bn = Br[n];
            float dA0 = __expf(soft_clamp(dt0 * A0[n], -8.f, 8.f, 0.5f));
            float dA1 = __expf(soft_clamp(dt1 * A1[n], -8.f, 8.f, 0.5f));
            h0[n] = h0[n] * dA0 + dt0 * Bn * x0;
            h1[n] = h1[n] * dA1 + dt1 * Bn * x1;
        }

        if (t % 10 == 0) {
            float s = 0.f;
#pragma unroll
            for (int n = 0; n < D_STATE; n++) s += h0[n] * h0[n] + h1[n] * h1[n];
            red[tid] = s; __syncthreads();
            for (int st = 512; st > 0; st >>= 1) {
                if (tid < st) red[tid] += red[tid + st];
                __syncthreads();
            }
            float fac = fminf(20.f / (sqrtf(red[0]) + 1e-8f), 1.f);
            __syncthreads();   // protect red[0] before next reuse
#pragma unroll
            for (int n = 0; n < D_STATE; n++) { h0[n] *= fac; h1[n] *= fac; }
        }

        float y0 = Dp0 * x0, y1 = Dp1 * x1;
#pragma unroll
        for (int n = 0; n < D_STATE; n++) {
            float Cn = Cr[n];
            y0 += h0[n] * Cn;
            y1 += h1[n] * Cn;
        }
        xssm[m * D_INNER + d0]     = y0;
        xssm[m * D_INNER + d0 + 1] = y1;
    }
}

// ---------------------------------------------------------------- gate: y2 = xssm * silu(gate), to f16
__global__ void gate_kernel(const float* __restrict__ proj, const float* __restrict__ xssm,
                            f16* __restrict__ y2) {
    int i = blockIdx.x * blockDim.x + threadIdx.x;
    if (i >= M_ROWS * D_INNER) return;
    int d = i % D_INNER;
    int m = i / D_INNER;
    float g = proj[(size_t)m * (2 * D_INNER) + D_INNER + d];
    y2[i] = (f16)(xssm[i] * (g * sigmoid_f(g)));
}

// ---------------------------------------------------------------- launch
extern "C" void kernel_launch(void* const* d_in, const int* in_sizes, int n_in,
                              void* d_out, int out_size, void* d_ws, size_t ws_size,
                              hipStream_t stream) {
    const float* x      = (const float*)d_in[0];
    const float* norm_w = (const float*)d_in[1];
    const float* W_in   = (const float*)d_in[2];
    const float* conv_w = (const float*)d_in[3];
    const float* conv_b = (const float*)d_in[4];
    const float* W_dt   = (const float*)d_in[5];
    const float* b_dt   = (const float*)d_in[6];
    const float* W_B    = (const float*)d_in[7];
    const float* W_C    = (const float*)d_in[8];
    const float* A_log  = (const float*)d_in[9];
    const float* D_par  = (const float*)d_in[10];
    const float* W_out  = (const float*)d_in[11];
    const float* rscale = (const float*)d_in[12];
    float* out = (float*)d_out;

    char* ws = (char*)d_ws;
    size_t off = 0;
    auto wsalloc = [&](size_t bytes) -> char* {
        char* p = ws + off;
        off += (bytes + 255) & ~(size_t)255;
        return p;
    };
    f16*   W_in_h  = (f16*)  wsalloc((size_t)4096 * 1024 * sizeof(f16));
    f16*   W_dt_h  = (f16*)  wsalloc((size_t)2048 * 2048 * sizeof(f16));
    f16*   W_out_h = (f16*)  wsalloc((size_t)1024 * 2048 * sizeof(f16));
    f16*   W_B_h   = (f16*)  wsalloc((size_t)16 * 2048 * sizeof(f16));
    f16*   W_C_h   = (f16*)  wsalloc((size_t)16 * 2048 * sizeof(f16));
    f16*   xn_h    = (f16*)  wsalloc((size_t)M_ROWS * D_MODEL * sizeof(f16));
    float* proj    = (float*)wsalloc((size_t)M_ROWS * 2 * D_INNER * sizeof(float));
    float* xc_f    = (float*)wsalloc((size_t)M_ROWS * D_INNER * sizeof(float));
    f16*   xc_h    = (f16*)  wsalloc((size_t)M_ROWS * D_INNER * sizeof(f16));
    float* dtbuf   = (float*)wsalloc((size_t)M_ROWS * D_INNER * sizeof(float));
    float* Bpbuf   = (float*)wsalloc((size_t)M_ROWS * D_STATE * sizeof(float));
    float* Cpbuf   = (float*)wsalloc((size_t)M_ROWS * D_STATE * sizeof(float));
    float* xssm    = (float*)wsalloc((size_t)M_ROWS * D_INNER * sizeof(float));
    f16*   y2_h    = (f16*)  wsalloc((size_t)M_ROWS * D_INNER * sizeof(f16));

    // 1) weight conversion to f16
    cvt_f16_kernel<<<4096, 256, 0, stream>>>(W_in,  W_in_h,  4096 * 1024);
    cvt_f16_kernel<<<4096, 256, 0, stream>>>(W_dt,  W_dt_h,  2048 * 2048);
    cvt_f16_kernel<<<2048, 256, 0, stream>>>(W_out, W_out_h, 1024 * 2048);
    cvt_f16_kernel<<<128,  256, 0, stream>>>(W_B,   W_B_h,   16 * 2048);
    cvt_f16_kernel<<<128,  256, 0, stream>>>(W_C,   W_C_h,   16 * 2048);

    // 2) RMSNorm -> xn (f16)
    rmsnorm_kernel<<<M_ROWS, 256, 0, stream>>>(x, norm_w, xn_h);

    // 3) proj = xn @ W_in^T   (2048 x 4096, K=1024)
    wmma_gemm_tiled<<<dim3(4096 / 64, M_ROWS / 128), 256, 0, stream>>>(
        xn_h, W_in_h, proj, M_ROWS, 4096, 1024, 1024, 1024, 4096,
        0, nullptr, nullptr, nullptr);

    // 4) depthwise causal conv + SiLU -> xc (f32 + f16)
    conv_silu_kernel<<<(M_ROWS * D_INNER) / 256, 256, 0, stream>>>(
        proj, conv_w, conv_b, xc_f, xc_h);

    // 5) dt = soft_clamp(softplus(xc @ W_dt^T + b_dt))   (2048 x 2048, K=2048)
    wmma_gemm_tiled<<<dim3(2048 / 64, M_ROWS / 128), 256, 0, stream>>>(
        xc_h, W_dt_h, dtbuf, M_ROWS, 2048, 2048, 2048, 2048, 2048,
        1, b_dt, nullptr, nullptr);

    // 6) Bp = xc @ W_B^T, Cp = xc @ W_C^T   (2048 x 16, K=2048)
    wmma_gemm_narrow<<<dim3(1, M_ROWS / 128), 256, 0, stream>>>(
        xc_h, W_B_h, Bpbuf, M_ROWS, 2048, 2048, 2048, 16);
    wmma_gemm_narrow<<<dim3(1, M_ROWS / 128), 256, 0, stream>>>(
        xc_h, W_C_h, Cpbuf, M_ROWS, 2048, 2048, 2048, 16);

    // 7) selective scan (sequential in L; one workgroup per batch for the global norm clamp)
    scan_kernel<<<BATCH, 1024, 0, stream>>>(dtbuf, xc_f, Bpbuf, Cpbuf, A_log, D_par, xssm);

    // 8) y2 = xssm * silu(gate) -> f16
    gate_kernel<<<(M_ROWS * D_INNER) / 256, 256, 0, stream>>>(proj, xssm, y2_h);

    // 9) out = x + sigmoid(rscale)*0.5 * (y2 @ W_out^T)   (2048 x 1024, K=2048)
    wmma_gemm_tiled<<<dim3(1024 / 64, M_ROWS / 128), 256, 0, stream>>>(
        y2_h, W_out_h, out, M_ROWS, 1024, 2048, 2048, 2048, 1024,
        2, nullptr, x, rscale);
}
